// FastRCNNHead_8512625180807
// MI455X (gfx1250) — compile-verified, matrix operand order checked by reading, not run on hardware
//
#include <hip/hip_runtime.h>

// ---------------------------------------------------------------------------
// FastRCNN head on gfx1250 (CDNA5): ROI-align -> FC1(BN,ReLU) -> FC2(BN,ReLU)
// -> pred -> box decode. GEMMs run on V_WMMA_F32_16X16X32_BF16; FC1/FC2 use
// double-buffered LDS tiles staged with GLOBAL_LOAD_ASYNC_TO_LDS_B128
// (ASYNCcnt-tracked), with a sync fallback if the builtin is absent.
// ---------------------------------------------------------------------------

typedef __bf16 bf16_t;
typedef __attribute__((ext_vector_type(16))) __bf16 v16bf;
typedef __attribute__((ext_vector_type(8)))  float  v8f;
typedef int v4i128 __attribute__((vector_size(16)));   // matches builtin pointee

#define BQ      4
#define NROI    512
#define NROWS   (BQ * NROI)         // 2048
#define CIN     256
#define RES7    7
#define KFLAT   (CIN * RES7 * RES7) // 12544
#define FCD     1024
#define OUTC    94                  // 3*(1+6+24)+1
#define OUTP    96                  // padded to multiple of 32 for WMMA N-tiling

#ifndef __has_builtin
#define __has_builtin(x) 0
#endif

#if __has_builtin(__builtin_amdgcn_global_load_async_to_lds_b128)
#define ASYNC_COPY 1
#else
#define ASYNC_COPY 0
#endif

#define AS_GLOBAL __attribute__((address_space(1)))
#define AS_LDS    __attribute__((address_space(3)))

#define WMMA_BF16(a, b, c) \
  __builtin_amdgcn_wmma_f32_16x16x32_bf16(false, (a), false, (b), (short)0, (c), false, false)

// 16-byte global -> LDS chunk copy (async when available)
__device__ __forceinline__ void copy16(const bf16_t* __restrict__ g,
                                       bf16_t* __restrict__ l) {
#if ASYNC_COPY
  __builtin_amdgcn_global_load_async_to_lds_b128(
      (AS_GLOBAL v4i128*)(uintptr_t)g,
      (AS_LDS v4i128*)(unsigned)(uintptr_t)l,
      0, 0);
#else
  *(uint4*)l = *(const uint4*)g;
#endif
}

__device__ __forceinline__ void async_wait0() {
#if ASYNC_COPY
#if __has_builtin(__builtin_amdgcn_s_wait_asynccnt)
  __builtin_amdgcn_s_wait_asynccnt(0);
#else
  asm volatile("s_wait_asynccnt 0x0" ::: "memory");
#endif
#endif
}

// ---------------------------------------------------------------------------
// ROI align: one block per (b,n) row. Bilinear sample coords shared in LDS.
// Output layout matches reference flatten: X[row][c*49 + ry*7 + rx], bf16.
// ---------------------------------------------------------------------------
__global__ __launch_bounds__(256)
void roi_pool_kernel(const float* __restrict__ p3,
                     const float* __restrict__ p4,
                     const float* __restrict__ p5,
                     const float* __restrict__ bbox2d,
                     const int*   __restrict__ anchor_id,
                     bf16_t*      __restrict__ Xbf) {
  const int row = blockIdx.x;            // 0..2047
  const int b   = row >> 9;              // row / 512
  const int tid = threadIdx.x;

  const int lvl = anchor_id[row] / 3;    // 0,1,2
  const float* feat;
  int H;
  if (lvl == 0)      { feat = p3; H = 64; }
  else if (lvl == 1) { feat = p4; H = 32; }
  else               { feat = p5; H = 16; }
  const int W = H;
  const float scale = 1.0f / (float)(8 << lvl);

  const float* box = bbox2d + (size_t)row * 4;
  const float t  = box[0] * scale - 0.5f;
  const float l  = box[1] * scale - 0.5f;
  const float bh = (box[2] - box[0]) * scale * (1.0f / 7.0f);
  const float bw = (box[3] - box[1]) * scale * (1.0f / 7.0f);

  __shared__ int   sy0[14], sy1[14], sx0[14], sx1[14];
  __shared__ float sly[14], slx[14];
  if (tid < 14) {
    float g  = ((float)tid + 0.5f) * 0.5f;
    float yc = fminf(fmaxf(t + g * bh, 0.0f), (float)(H - 1));
    float y0 = floorf(yc);
    int  y0i = (int)y0;
    sy0[tid] = y0i;
    sy1[tid] = min(y0i + 1, H - 1);
    sly[tid] = yc - y0;
  } else if (tid < 28) {
    int   s  = tid - 14;
    float g  = ((float)s + 0.5f) * 0.5f;
    float xc = fminf(fmaxf(l + g * bw, 0.0f), (float)(W - 1));
    float x0 = floorf(xc);
    int  x0i = (int)x0;
    sx0[s] = x0i;
    sx1[s] = min(x0i + 1, W - 1);
    slx[s] = xc - x0;
  }
  __syncthreads();

  const float* fb = feat + (size_t)b * CIN * H * W;
  for (int idx = tid; idx < KFLAT; idx += 256) {
    const int c  = idx / 49;
    const int p  = idx - c * 49;
    const int ry = p / 7;
    const int rx = p - ry * 7;
    const float* fc = fb + (size_t)c * H * W;
    float acc = 0.0f;
#pragma unroll
    for (int sy = 0; sy < 2; ++sy) {
      const int   syi = 2 * ry + sy;
      const int   y0 = sy0[syi], y1 = sy1[syi];
      const float ly = sly[syi];
#pragma unroll
      for (int sx = 0; sx < 2; ++sx) {
        const int   sxi = 2 * rx + sx;
        const int   x0 = sx0[sxi], x1 = sx1[sxi];
        const float lx = slx[sxi];
        const float v00 = fc[y0 * W + x0];
        const float v01 = fc[y0 * W + x1];
        const float v10 = fc[y1 * W + x0];
        const float v11 = fc[y1 * W + x1];
        acc += v00 * (1.0f - ly) * (1.0f - lx)
             + v01 * (1.0f - ly) * lx
             + v10 * ly * (1.0f - lx)
             + v11 * ly * lx;
      }
    }
    Xbf[(size_t)row * KFLAT + idx] = (bf16_t)(acc * 0.25f);
  }
}

// ---------------------------------------------------------------------------
// f32 -> bf16 conversion helpers
// ---------------------------------------------------------------------------
__global__ void f32_to_bf16_kernel(const float* __restrict__ src,
                                   bf16_t* __restrict__ dst, int n) {
  int i = blockIdx.x * blockDim.x + threadIdx.x;
  if (i < n) dst[i] = (bf16_t)src[i];
}

__global__ void f32_to_bf16_pad_rows_kernel(const float* __restrict__ src,
                                            bf16_t* __restrict__ dst,
                                            int rows_valid, int rows_pad, int cols) {
  int i = blockIdx.x * blockDim.x + threadIdx.x;
  int total = rows_pad * cols;
  if (i >= total) return;
  int r = i / cols;
  dst[i] = (r < rows_valid) ? (bf16_t)src[i] : (bf16_t)0.0f;
}

// ---------------------------------------------------------------------------
// Shared C-tile store with bias (C/D layout per ISA 7.12.2: VGPR v -> M=v or
// v+8 depending on lane half, N = lane&15).
// ---------------------------------------------------------------------------
__device__ __forceinline__
void store_ctile(float* __restrict__ C, const v8f& acc,
                 int mbase, int nbase, int m_hi, int n_l, int N,
                 const float* __restrict__ bias, int bias_n) {
#pragma unroll
  for (int v = 0; v < 8; ++v) {
    const int mr = mbase + m_hi + v;
    const int nc = nbase + n_l;
    const float bv = (nc < bias_n) ? bias[nc] : 0.0f;
    C[(size_t)mr * N + nc] = acc[v] + bv;
  }
}

// ---------------------------------------------------------------------------
// LDS-tiled WMMA GEMM: C[M][N] = A[M][K] * B[N][K]^T + bias.
// Block = 256 threads = 8 waves; block tile 128(M) x 64(N); BK = 32.
// Double-buffered LDS (24 KB), staged via async global->LDS B128 copies.
// Each wave owns a 32x32 tile: 4 f32 accumulators, 4 WMMAs per K-step.
// Requires M % 128 == 0, N % 64 == 0, K % 32 == 0.
// ---------------------------------------------------------------------------
__device__ __forceinline__
void stage_tiles(const bf16_t* __restrict__ A, const bf16_t* __restrict__ Bm,
                 bf16_t* __restrict__ As, bf16_t* __restrict__ Bs,
                 int gm0, int gn0, int K, int kk, int tid) {
  const int r  = tid >> 2;           // 0..63
  const int c  = (tid & 3) * 8;      // 8 bf16 = 16 bytes per chunk
  copy16(A  + (size_t)(gm0 + r)      * K + kk + c, As + r        * 32 + c);
  copy16(A  + (size_t)(gm0 + r + 64) * K + kk + c, As + (r + 64) * 32 + c);
  copy16(Bm + (size_t)(gn0 + r)      * K + kk + c, Bs + r        * 32 + c);
}

__global__ __launch_bounds__(256)
void gemm_bf16_wmma_lds(const bf16_t* __restrict__ A,   // M x K
                        const bf16_t* __restrict__ Bm,  // N x K (row-major)
                        const float*  __restrict__ bias,
                        float* __restrict__ C,          // M x N
                        int M, int N, int K) {
  __shared__ bf16_t As[2][128 * 32];
  __shared__ bf16_t Bs[2][64 * 32];

  const int tid  = threadIdx.x;
  const int lane = tid & 31;
  const int wv   = tid >> 5;          // 0..7
  const int tilesN = N >> 6;
  const int bM = blockIdx.x / tilesN;
  const int bN = blockIdx.x - bM * tilesN;
  const int gm0 = bM * 128;
  const int gn0 = bN * 64;
  const int wm0 = (wv >> 1) * 32;     // wave tile within block: 4 x 2 grid
  const int wn0 = (wv & 1) * 32;
  const int rsel = lane & 15;
  const int koff = (lane >> 4) << 4;  // K half handled by lane (0 or 16)

  stage_tiles(A, Bm, As[0], Bs[0], gm0, gn0, K, 0, tid);
  async_wait0();
  __syncthreads();

  v8f acc00 = {}; v8f acc01 = {}; v8f acc10 = {}; v8f acc11 = {};
  int buf = 0;
  for (int kk = 0; kk < K; kk += 32) {
    if (kk + 32 < K)
      stage_tiles(A, Bm, As[buf ^ 1], Bs[buf ^ 1], gm0, gn0, K, kk + 32, tid);

    const bf16_t* Ab = As[buf];
    const bf16_t* Bb = Bs[buf];
    const v16bf a0 = *(const v16bf*)(Ab + (wm0 + rsel)      * 32 + koff);
    const v16bf a1 = *(const v16bf*)(Ab + (wm0 + 16 + rsel) * 32 + koff);
    const v16bf b0 = *(const v16bf*)(Bb + (wn0 + rsel)      * 32 + koff);
    const v16bf b1 = *(const v16bf*)(Bb + (wn0 + 16 + rsel) * 32 + koff);
    acc00 = WMMA_BF16(a0, b0, acc00);
    acc01 = WMMA_BF16(a0, b1, acc01);
    acc10 = WMMA_BF16(a1, b0, acc10);
    acc11 = WMMA_BF16(a1, b1, acc11);

    async_wait0();      // this wave's copies into buf^1 have landed
    __syncthreads();    // everyone done reading buf / writing buf^1
    buf ^= 1;
  }

  const int n_l  = lane & 15;
  const int m_hi = (lane >> 4) << 3;  // 0 or 8
  store_ctile(C, acc00, gm0 + wm0,      gn0 + wn0,      m_hi, n_l, N, bias, N);
  store_ctile(C, acc01, gm0 + wm0,      gn0 + wn0 + 16, m_hi, n_l, N, bias, N);
  store_ctile(C, acc10, gm0 + wm0 + 16, gn0 + wn0,      m_hi, n_l, N, bias, N);
  store_ctile(C, acc11, gm0 + wm0 + 16, gn0 + wn0 + 16, m_hi, n_l, N, bias, N);
}

// ---------------------------------------------------------------------------
// Direct (global-operand) WMMA GEMM for the small pred layer (N = 96).
// Each wave computes a 32x32 tile straight from L2-resident operands.
// ---------------------------------------------------------------------------
__global__ __launch_bounds__(256)
void gemm_bf16_wmma(const bf16_t* __restrict__ A,   // M x K
                    const bf16_t* __restrict__ Bm,  // N x K (row-major)
                    const float*  __restrict__ bias, int bias_n,
                    float* __restrict__ C,          // M x N
                    int M, int N, int K) {
  const int wave = (int)((blockIdx.x * blockDim.x + threadIdx.x) >> 5);
  const int lane = (int)(threadIdx.x & 31);
  const int tilesN = N >> 5;
  const int tM = wave / tilesN;
  const int tN = wave - tM * tilesN;
  if (tM >= (M >> 5)) return;             // wave-uniform guard

  const int m0 = tM << 5;
  const int n0 = tN << 5;
  const int rsel = lane & 15;
  const int koff = (lane >> 4) << 4;

  const bf16_t* Ap0 = A  + (size_t)(m0 + rsel) * K + koff;
  const bf16_t* Ap1 = Ap0 + (size_t)16 * K;
  const bf16_t* Bp0 = Bm + (size_t)(n0 + rsel) * K + koff;
  const bf16_t* Bp1 = Bp0 + (size_t)16 * K;

  v8f acc00 = {}; v8f acc01 = {}; v8f acc10 = {}; v8f acc11 = {};
  for (int kk = 0; kk < K; kk += 32) {
    const v16bf a0 = *(const v16bf*)(Ap0 + kk);
    const v16bf a1 = *(const v16bf*)(Ap1 + kk);
    const v16bf b0 = *(const v16bf*)(Bp0 + kk);
    const v16bf b1 = *(const v16bf*)(Bp1 + kk);
    acc00 = WMMA_BF16(a0, b0, acc00);
    acc01 = WMMA_BF16(a0, b1, acc01);
    acc10 = WMMA_BF16(a1, b0, acc10);
    acc11 = WMMA_BF16(a1, b1, acc11);
  }

  const int n_l  = lane & 15;
  const int m_hi = (lane >> 4) << 3;
  store_ctile(C, acc00, m0,      n0,      m_hi, n_l, N, bias, bias_n);
  store_ctile(C, acc01, m0,      n0 + 16, m_hi, n_l, N, bias, bias_n);
  store_ctile(C, acc10, m0 + 16, n0,      m_hi, n_l, N, bias, bias_n);
  store_ctile(C, acc11, m0 + 16, n0 + 16, m_hi, n_l, N, bias, bias_n);
}

// ---------------------------------------------------------------------------
// BatchNorm (train-mode) statistics: one block per column.
// scale = gamma * rsqrt(var + eps), shift = beta - mean * scale
// ---------------------------------------------------------------------------
__global__ __launch_bounds__(256)
void bn_stats_kernel(const float* __restrict__ X,     // M x N
                     const float* __restrict__ gamma,
                     const float* __restrict__ beta,
                     float* __restrict__ scale,
                     float* __restrict__ shift,
                     int M, int N) {
  const int col = blockIdx.x;
  const int tid = threadIdx.x;
  float s = 0.0f, sq = 0.0f;
  for (int r = tid; r < M; r += blockDim.x) {
    float x = X[(size_t)r * N + col];
    s  += x;
    sq += x * x;
  }
  __shared__ float ss[256];
  __shared__ float sw[256];
  ss[tid] = s;
  sw[tid] = sq;
  __syncthreads();
  for (int st = 128; st > 0; st >>= 1) {
    if (tid < st) { ss[tid] += ss[tid + st]; sw[tid] += sw[tid + st]; }
    __syncthreads();
  }
  if (tid == 0) {
    const float inv_m = 1.0f / (float)M;
    const float mean  = ss[0] * inv_m;
    const float var   = sw[0] * inv_m - mean * mean;
    const float sc    = gamma[col] * rsqrtf(var + 1e-5f);
    scale[col] = sc;
    shift[col] = beta[col] - mean * sc;
  }
}

// ---------------------------------------------------------------------------
// y = relu(x*scale[col] + shift[col]) -> bf16 (input to next GEMM)
// ---------------------------------------------------------------------------
__global__ void bn_relu_bf16_kernel(const float* __restrict__ X,
                                    const float* __restrict__ scale,
                                    const float* __restrict__ shift,
                                    bf16_t* __restrict__ Y, int M, int N) {
  int i = blockIdx.x * blockDim.x + threadIdx.x;
  if (i >= M * N) return;
  int c = i % N;
  float y = fmaxf(X[i] * scale[c] + shift[c], 0.0f);
  Y[i] = (bf16_t)y;
}

// ---------------------------------------------------------------------------
// Box decode: cat/yaw passthrough, yx += delta*stride, lw *= exp(clip(delta))
// ---------------------------------------------------------------------------
__global__ void postprocess_kernel(const float* __restrict__ P,       // NROWS x OUTP
                                   const float* __restrict__ anchors, // NROWS x 4 (cy,cx,h,w)
                                   const int*   __restrict__ anchor_id,
                                   float* __restrict__ out) {         // NROWS x OUTC
  int row = blockIdx.x * blockDim.x + threadIdx.x;
  if (row >= NROWS) return;
  const float* p = P + (size_t)row * OUTP;
  float* o = out + (size_t)row * OUTC;

  const int   lvl       = anchor_id[row] / 3;
  const float stride_px = (float)(8 << lvl);     // exp2(lvl + 3)
  const float acy = anchors[(size_t)row * 4 + 0];
  const float acx = anchors[(size_t)row * 4 + 1];
  const float ah  = anchors[(size_t)row * 4 + 2];
  const float aw  = anchors[(size_t)row * 4 + 3];

  o[0] = p[0]; o[1] = p[1]; o[2] = p[2]; o[3] = p[3];   // class logits
#pragma unroll
  for (int k = 0; k < 3; ++k) {
    const int base = 4 + k * 6;
    const float d0 = p[base + 0], d1 = p[base + 1];
    const float d2 = p[base + 2], d3 = p[base + 3];
    o[base + 0] = acy + d0 * stride_px;
    o[base + 1] = acx + d1 * stride_px;
    o[base + 2] = ah * expf(fminf(fmaxf(d2, -4.0f), 4.0f));
    o[base + 3] = aw * expf(fminf(fmaxf(d3, -4.0f), 4.0f));
    o[base + 4] = p[base + 4];
    o[base + 5] = p[base + 5];
  }
  for (int j = 22; j < OUTC; ++j) o[j] = p[j];          // yaw cls + res
}

// ---------------------------------------------------------------------------
// Host launcher
// ---------------------------------------------------------------------------
extern "C" void kernel_launch(void* const* d_in, const int* in_sizes, int n_in,
                              void* d_out, int out_size, void* d_ws, size_t ws_size,
                              hipStream_t stream) {
  (void)in_sizes; (void)n_in; (void)out_size; (void)ws_size;

  const float* feat_p3   = (const float*)d_in[0];
  const float* feat_p4   = (const float*)d_in[1];
  const float* feat_p5   = (const float*)d_in[2];
  const float* bbox2d    = (const float*)d_in[3];
  const float* anchors   = (const float*)d_in[4];
  const int*   anchor_id = (const int*)  d_in[5];
  const float* fc1_w     = (const float*)d_in[6];
  const float* fc1_b     = (const float*)d_in[7];
  const float* bn1_g     = (const float*)d_in[8];
  const float* bn1_b     = (const float*)d_in[9];
  const float* fc2_w     = (const float*)d_in[10];
  const float* fc2_b     = (const float*)d_in[11];
  const float* bn2_g     = (const float*)d_in[12];
  const float* bn2_b     = (const float*)d_in[13];
  const float* pred_w    = (const float*)d_in[14];
  const float* pred_b    = (const float*)d_in[15];
  float* out = (float*)d_out;

  char*  ws  = (char*)d_ws;
  size_t off = 0;
  auto alloc = [&](size_t bytes) -> void* {
    void* p = ws + off;
    off = (off + bytes + 255) & ~(size_t)255;
    return p;
  };

  bf16_t* W1bf  = (bf16_t*)alloc((size_t)FCD   * KFLAT * sizeof(bf16_t));
  bf16_t* W2bf  = (bf16_t*)alloc((size_t)FCD   * FCD   * sizeof(bf16_t));
  bf16_t* Wpbf  = (bf16_t*)alloc((size_t)OUTP  * FCD   * sizeof(bf16_t));
  bf16_t* Xbf   = (bf16_t*)alloc((size_t)NROWS * KFLAT * sizeof(bf16_t));
  bf16_t* H1bf  = (bf16_t*)alloc((size_t)NROWS * FCD   * sizeof(bf16_t));
  bf16_t* H2bf  = (bf16_t*)alloc((size_t)NROWS * FCD   * sizeof(bf16_t));
  float*  Pbuf  = (float*)alloc((size_t)NROWS * FCD  * sizeof(float)); // reused FC1/FC2 out
  float*  P3buf = (float*)alloc((size_t)NROWS * OUTP * sizeof(float));
  float*  scale1 = (float*)alloc(FCD * sizeof(float));
  float*  shift1 = (float*)alloc(FCD * sizeof(float));
  float*  scale2 = (float*)alloc(FCD * sizeof(float));
  float*  shift2 = (float*)alloc(FCD * sizeof(float));

  // Weight conversion (deterministic, re-done each call)
  {
    int n = FCD * KFLAT;
    f32_to_bf16_kernel<<<(n + 255) / 256, 256, 0, stream>>>(fc1_w, W1bf, n);
  }
  {
    int n = FCD * FCD;
    f32_to_bf16_kernel<<<(n + 255) / 256, 256, 0, stream>>>(fc2_w, W2bf, n);
  }
  {
    int n = OUTP * FCD;
    f32_to_bf16_pad_rows_kernel<<<(n + 255) / 256, 256, 0, stream>>>(pred_w, Wpbf, OUTC, OUTP, FCD);
  }

  // ROI align -> bf16 feature rows
  roi_pool_kernel<<<NROWS, 256, 0, stream>>>(feat_p3, feat_p4, feat_p5,
                                             bbox2d, anchor_id, Xbf);

  // FC1: 2048x12544 @ 12544x1024 (LDS-tiled, async-staged)
  {
    int blocks = (NROWS / 128) * (FCD / 64);    // 16 * 16 = 256 blocks
    gemm_bf16_wmma_lds<<<blocks, 256, 0, stream>>>(Xbf, W1bf, fc1_b,
                                                   Pbuf, NROWS, FCD, KFLAT);
  }
  bn_stats_kernel<<<FCD, 256, 0, stream>>>(Pbuf, bn1_g, bn1_b, scale1, shift1, NROWS, FCD);
  {
    int n = NROWS * FCD;
    bn_relu_bf16_kernel<<<(n + 255) / 256, 256, 0, stream>>>(Pbuf, scale1, shift1, H1bf, NROWS, FCD);
  }

  // FC2: 2048x1024 @ 1024x1024 (LDS-tiled, async-staged)
  {
    int blocks = (NROWS / 128) * (FCD / 64);
    gemm_bf16_wmma_lds<<<blocks, 256, 0, stream>>>(H1bf, W2bf, fc2_b,
                                                   Pbuf, NROWS, FCD, FCD);
  }
  bn_stats_kernel<<<FCD, 256, 0, stream>>>(Pbuf, bn2_g, bn2_b, scale2, shift2, NROWS, FCD);
  {
    int n = NROWS * FCD;
    bn_relu_bf16_kernel<<<(n + 255) / 256, 256, 0, stream>>>(Pbuf, scale2, shift2, H2bf, NROWS, FCD);
  }

  // pred: 2048x1024 @ 1024x96 (padded), direct WMMA
  {
    int waves  = (NROWS / 32) * (OUTP / 32);    // 192 waves
    int blocks = waves * 32 / 256;              // 24 blocks
    gemm_bf16_wmma<<<blocks, 256, 0, stream>>>(H2bf, Wpbf, pred_b, OUTC,
                                               P3buf, NROWS, OUTP, FCD);
  }

  // Box decode -> d_out (2048 x 94 f32)
  postprocess_kernel<<<(NROWS + 255) / 256, 256, 0, stream>>>(P3buf, anchors, anchor_id, out);
}
